// ImprovedGATPredictor_77197742178682
// MI455X (gfx1250) — matrix operand order repaired and checked by reading, not compiled, not run on hardware
//
#include <hip/hip_runtime.h>
#include <cstdint>

#define NN   50000
#define EE   800000
#define HIDD 128
#define EPSV 1e-5f

typedef __attribute__((ext_vector_type(16))) _Float16 v16h;
typedef __attribute__((ext_vector_type(8)))  float    v8f;

// exact pointee type clang expects for the async-LDS builtins: int __vector(4)
typedef int v4i __attribute__((vector_size(16)));
typedef __attribute__((address_space(1))) v4i* gv4i_t;   // global
typedef __attribute__((address_space(3))) v4i* lv4i_t;   // LDS

#if defined(__HIP_DEVICE_COMPILE__) && \
    __has_builtin(__builtin_amdgcn_global_load_async_to_lds_b128) && \
    __has_builtin(__builtin_amdgcn_s_wait_asynccnt)
#define HAS_ASYNC_LDS 1
#else
#define HAS_ASYNC_LDS 0
#endif

// ---------------- helpers ----------------
__device__ __forceinline__ unsigned fenc(float f) {
  unsigned u = __float_as_uint(f);
  return (u & 0x80000000u) ? ~u : (u | 0x80000000u);
}
__device__ __forceinline__ float fdec(unsigned u) {
  return __uint_as_float((u & 0x80000000u) ? (u & 0x7fffffffu) : ~u);
}

// ---------------- weight convert: W[K,128] f32 -> Wt[128,K] f16 (transposed) ------
__global__ void k_convert_wt(const float* __restrict__ W, _Float16* __restrict__ Wt, int K) {
  int t = blockIdx.x * blockDim.x + threadIdx.x;
  if (t >= 128 * K) return;
  int n = t / K, k = t - n * K;
  Wt[t] = (_Float16)W[k * 128 + n];
}

// ---------------- GEMM: C[M,128] = act(A[M,KD] @ W + bias), WMMA f16->f32 --------
// block: 256 thr (8 waves). Block tile: 64 rows x 128 cols. Wave w -> cols [16w,16w+16).
// K staged in 64-wide LDS chunks, double-buffered; W^T chunks staged with
// GLOBAL_LOAD_ASYNC_TO_LDS_B128 (ASYNCcnt) so staging overlaps the WMMA burst.
// Padded LDS row stride (72 halves) -> conflict-free ds_load_b128.
template<int KD, bool RELU>
__global__ __launch_bounds__(256) void k_gemm(const float* __restrict__ A,
                                              const _Float16* __restrict__ Wt,
                                              const float* __restrict__ bias,
                                              float* __restrict__ C, int M) {
  static_assert(KD % 64 == 0, "KD multiple of 64");
  constexpr int LDK = 64 + 8;                 // halves per LDS row
  constexpr int NCH = KD / 64;                // K chunks
  __shared__ _Float16 As[2][64 * LDK];        // 2 x 9.2 KB
  __shared__ _Float16 Ws[2][128 * LDK];       // 2 x 18.4 KB
  const int t    = threadIdx.x;
  const int m0   = blockIdx.x * 64;
  const int wave = t >> 5, lane = t & 31;
  const int n16  = lane & 15, hi = lane >> 4;
  const int ncol = wave * 16 + n16;

  auto stage = [&](int ci, int buf) {
    const int k0 = ci * 64;
    // W^T chunk: 128 rows x 64 halves (f16, pre-transposed in global)
    for (int idx = t; idx < 128 * 8; idx += 256) {
      int row = idx >> 3, ch = idx & 7;
      const _Float16* gp = Wt + (size_t)row * KD + k0 + ch * 8;
      _Float16* lp = &Ws[buf][row * LDK + ch * 8];
#if HAS_ASYNC_LDS
      __builtin_amdgcn_global_load_async_to_lds_b128((gv4i_t)gp, (lv4i_t)lp, 0, 0);
#else
      *(uint4*)lp = *(const uint4*)gp;
#endif
    }
    // A chunk: 64 rows x 64 floats, f32 -> f16 through registers
    for (int idx = t; idx < 64 * 16; idx += 256) {
      int row = idx >> 4, ch = idx & 15;
      int gr = m0 + row;
      float4 v = {0.f, 0.f, 0.f, 0.f};
      if (gr < M) v = *(const float4*)(A + (size_t)gr * KD + k0 + ch * 4);
      union { _Float16 h[4]; uint2 u; } cv;
      cv.h[0] = (_Float16)v.x; cv.h[1] = (_Float16)v.y;
      cv.h[2] = (_Float16)v.z; cv.h[3] = (_Float16)v.w;
      *(uint2*)(&As[buf][row * LDK + ch * 4]) = cv.u;
    }
  };

  const float bv_ = bias[ncol];
  v8f acc[4];
  for (int mt = 0; mt < 4; ++mt) { v8f z; for (int j = 0; j < 8; ++j) z[j] = bv_; acc[mt] = z; }

  stage(0, 0);
#if HAS_ASYNC_LDS
  __builtin_amdgcn_s_wait_asynccnt(0);
#endif
  __syncthreads();

  for (int c = 0; c < NCH; ++c) {
    const int buf = c & 1;
    if (c + 1 < NCH) stage(c + 1, buf ^ 1);   // prefetch next chunk (async)
    for (int kk = 0; kk < 64; kk += 32) {
      // B fragment (16-bit B 32x16, contiguous K order used here): lane N=L%16, VGPR j holds K=16*hi+2j,2j+1
      union { uint4 u[2]; v16h h; } bf;
      const _Float16* wp = &Ws[buf][ncol * LDK + kk + 16 * hi];
      bf.u[0] = *(const uint4*)(wp);
      bf.u[1] = *(const uint4*)(wp + 8);
      for (int mt = 0; mt < 4; ++mt) {
        // A fragment (16-bit A 16x32): lane M=L%16, VGPR j holds K=16*hi+2j,2j+1 (matches B)
        union { uint4 u[2]; v16h h; } af;
        const _Float16* ap = &As[buf][(mt * 16 + n16) * LDK + kk + 16 * hi];
        af.u[0] = *(const uint4*)(ap);
        af.u[1] = *(const uint4*)(ap + 8);
        acc[mt] = __builtin_amdgcn_wmma_f32_16x16x32_f16(
            false, af.h, false, bf.h, (short)0, acc[mt], false, false);
      }
    }
    if (c + 1 < NCH) {
#if HAS_ASYNC_LDS
      __builtin_amdgcn_s_wait_asynccnt(0);
#endif
      __syncthreads();
    }
  }

  // D layout: VGPR j -> M = j + 8*hi, N = ncol
  for (int mt = 0; mt < 4; ++mt)
    for (int j = 0; j < 8; ++j) {
      int gr = m0 + mt * 16 + hi * 8 + j;
      if (gr < M) {
        float v = acc[mt][j];
        if (RELU) v = fmaxf(v, 0.f);
        C[(size_t)gr * HIDD + ncol] = v;
      }
    }
}

// ---------------- BatchNorm ----------------
__global__ void k_bn_zero(float* acc) { if (threadIdx.x < 256) acc[threadIdx.x] = 0.f; }

__global__ __launch_bounds__(256) void k_bn_stats(const float* __restrict__ X, int M,
                                                  float* __restrict__ acc) {
  int g = blockIdx.x * 256 + threadIdx.x;
  int c = g & 127;
  int r = g >> 7;
  int rstep = gridDim.x * 2;
  float s = 0.f, q = 0.f;
  for (; r < M; r += rstep) {
    float v = X[(size_t)r * HIDD + c];
    s += v; q += v * v;
  }
  atomicAdd(&acc[c], s);
  atomicAdd(&acc[128 + c], q);
}

__global__ void k_bn_finalize(const float* __restrict__ acc, float* __restrict__ stat, int M) {
  int c = threadIdx.x;
  if (c >= 128) return;
  float inv_m = 1.f / (float)M;
  float mean = acc[c] * inv_m;
  float var  = acc[128 + c] * inv_m - mean * mean;
  var = fmaxf(var, 0.f);
  stat[c]       = mean;
  stat[128 + c] = rsqrtf(var + EPSV);
}

__global__ __launch_bounds__(256) void k_bn_apply_relu(float* __restrict__ X,
                                                       const float* __restrict__ stat,
                                                       const float* __restrict__ gam,
                                                       const float* __restrict__ bet,
                                                       int M) {
  int t = blockIdx.x * 256 + threadIdx.x;
  if (t >= M * HIDD) return;
  int c = t & 127;
  float v = X[t];
  float y = gam[c] * (v - stat[c]) * stat[128 + c] + bet[c];
  X[t] = fmaxf(y, 0.f);
}

// ---------------- attention (segment softmax) ----------------
__global__ __launch_bounds__(256) void k_seg_init(unsigned* __restrict__ m_enc,
                                                  float* __restrict__ ssum,
                                                  float* __restrict__ asum) {
  int t = blockIdx.x * 256 + threadIdx.x;
  if (t >= NN * 4) return;
  m_enc[t] = fenc(-3.0e38f);
  ssum[t] = 0.f;
  asum[t] = 0.f;
}

// one edge per wave: lane = head*8 + i; coalesced 512B row reads; shuffle reduce.
__global__ __launch_bounds__(256) void k_scores(const int* __restrict__ src,
                                                const int* __restrict__ dst,
                                                const float* __restrict__ Q,
                                                const float* __restrict__ K,
                                                float* __restrict__ scores,
                                                unsigned* __restrict__ m_enc) {
  int lane = threadIdx.x & 31;
  int e = blockIdx.x * 8 + (threadIdx.x >> 5);
  if (e >= EE) return;
  int head = lane >> 3, i = lane & 7;
  int s = src[e], d = dst[e];
  const float4 q = *(const float4*)(Q + (size_t)s * HIDD + head * 32 + i * 4);
  const float4 k = *(const float4*)(K + (size_t)d * HIDD + head * 32 + i * 4);
  float p = q.x * k.x + q.y * k.y + q.z * k.z + q.w * k.w;
  p += __shfl_down(p, 4, 8);
  p += __shfl_down(p, 2, 8);
  p += __shfl_down(p, 1, 8);
  if (i == 0) {
    p *= 0.17677669529663687f;  // 1/sqrt(32)
    scores[(size_t)e * 4 + head] = p;
    atomicMax(&m_enc[d * 4 + head], fenc(p));
  }
}

__global__ __launch_bounds__(256) void k_exp_sum(const int* __restrict__ dst,
                                                 float* __restrict__ scores,
                                                 const unsigned* __restrict__ m_enc,
                                                 float* __restrict__ ssum) {
  int t = blockIdx.x * 256 + threadIdx.x;
  if (t >= EE * 4) return;
  int e = t >> 2, head = t & 3;
  int d = dst[e];
  float mx = fdec(m_enc[d * 4 + head]);
  float ex = __expf(scores[t] - mx);
  scores[t] = ex;
  atomicAdd(&ssum[d * 4 + head], ex);
}

__global__ __launch_bounds__(256) void k_norm_acc(const int* __restrict__ dst,
                                                  const float* __restrict__ scores,
                                                  const float* __restrict__ ssum,
                                                  float* __restrict__ asum) {
  int t = blockIdx.x * 256 + threadIdx.x;
  if (t >= EE * 4) return;
  int e = t >> 2, head = t & 3;
  int d = dst[e];
  float w = scores[t] / ssum[d * 4 + head];
  atomicAdd(&asum[d * 4 + head], w);
}

// out[n,c] = V[n,c] * asum[n,head(c)]   (exact: V[dst] constant within a segment)
__global__ __launch_bounds__(256) void k_scale_v(const float* __restrict__ V,
                                                 const float* __restrict__ asum,
                                                 float* __restrict__ out) {
  int t = blockIdx.x * 256 + threadIdx.x;
  if (t >= NN * HIDD) return;
  int n = t >> 7, c = t & 127, head = c >> 5;
  out[t] = V[t] * asum[n * 4 + head];
}

// ---------------- host launch ----------------
extern "C" void kernel_launch(void* const* d_in, const int* in_sizes, int n_in,
                              void* d_out, int out_size, void* d_ws, size_t ws_size,
                              hipStream_t stream) {
  (void)in_sizes; (void)n_in; (void)out_size; (void)ws_size;
  const float* x   = (const float*)d_in[0];
  const int*   ei  = (const int*)d_in[1];
  const float* W1  = (const float*)d_in[2];
  const float* b1  = (const float*)d_in[3];
  const float* g1  = (const float*)d_in[4];
  const float* be1 = (const float*)d_in[5];
  const float* W2  = (const float*)d_in[6];
  const float* b2  = (const float*)d_in[7];
  const float* g2  = (const float*)d_in[8];
  const float* be2 = (const float*)d_in[9];
  const float* Wq  = (const float*)d_in[10];
  const float* bq  = (const float*)d_in[11];
  const float* Wk  = (const float*)d_in[12];
  const float* bk  = (const float*)d_in[13];
  const float* Wv  = (const float*)d_in[14];
  const float* bv  = (const float*)d_in[15];
  const float* Wo  = (const float*)d_in[16];
  const float* bo  = (const float*)d_in[17];
  const int* src = ei;
  const int* dst = ei + EE;

  // workspace carve-up (256B aligned)
  char* ws = (char*)d_ws;
  size_t off = 0;
  auto carve = [&](size_t bytes) { char* p = ws + off; off += (bytes + 255) & ~(size_t)255; return p; };
  _Float16* wt_e1 = (_Float16*)carve(128 * 256 * 2);
  _Float16* wt_e2 = (_Float16*)carve(128 * 128 * 2);
  _Float16* wt_q[2], *wt_k[2], *wt_v[2], *wt_o[2];
  for (int l = 0; l < 2; ++l) {
    wt_q[l] = (_Float16*)carve(128 * 128 * 2);
    wt_k[l] = (_Float16*)carve(128 * 128 * 2);
    wt_v[l] = (_Float16*)carve(128 * 128 * 2);
    wt_o[l] = (_Float16*)carve(128 * 128 * 2);
  }
  float* hA     = (float*)carve((size_t)NN * HIDD * 4);
  float* hB     = (float*)carve((size_t)NN * HIDD * 4);
  float* Qb     = (float*)carve((size_t)NN * HIDD * 4);
  float* Kb     = (float*)carve((size_t)NN * HIDD * 4);
  float* Vb     = (float*)carve((size_t)NN * HIDD * 4);
  float* scores = (float*)carve((size_t)EE * 4 * 4);
  unsigned* m_enc = (unsigned*)carve((size_t)NN * 4 * 4);
  float* ssum   = (float*)carve((size_t)NN * 4 * 4);
  float* asum   = (float*)carve((size_t)NN * 4 * 4);
  float* bn_acc = (float*)carve(256 * 4);
  float* bn_st  = (float*)carve(256 * 4);

  const int CONV128 = (128 * 128 + 255) / 256;
  const int CONV256 = (128 * 256 + 255) / 256;
  const int GEMMG   = (NN + 63) / 64;
  const int ELEMG   = (NN * HIDD + 255) / 256;
  const int EDGEG4  = (EE * 4 + 255) / 256;
  const int EDGEW   = (EE + 7) / 8;
  const int SEGG    = (NN * 4 + 255) / 256;

  // convert + transpose weights to f16 once per launch
  k_convert_wt<<<CONV256, 256, 0, stream>>>(W1, wt_e1, 256);
  k_convert_wt<<<CONV128, 256, 0, stream>>>(W2, wt_e2, 128);
  for (int l = 0; l < 2; ++l) {
    k_convert_wt<<<CONV128, 256, 0, stream>>>(Wq + l * 128 * 128, wt_q[l], 128);
    k_convert_wt<<<CONV128, 256, 0, stream>>>(Wk + l * 128 * 128, wt_k[l], 128);
    k_convert_wt<<<CONV128, 256, 0, stream>>>(Wv + l * 128 * 128, wt_v[l], 128);
    k_convert_wt<<<CONV128, 256, 0, stream>>>(Wo + l * 128 * 128, wt_o[l], 128);
  }

  // ---- encoder ----
  k_gemm<256, false><<<GEMMG, 256, 0, stream>>>(x, wt_e1, b1, hA, NN);
  k_bn_zero<<<1, 256, 0, stream>>>(bn_acc);
  k_bn_stats<<<256, 256, 0, stream>>>(hA, NN, bn_acc);
  k_bn_finalize<<<1, 128, 0, stream>>>(bn_acc, bn_st, NN);
  k_bn_apply_relu<<<ELEMG, 256, 0, stream>>>(hA, bn_st, g1, be1, NN);

  k_gemm<128, false><<<GEMMG, 256, 0, stream>>>(hA, wt_e2, b2, hB, NN);
  k_bn_zero<<<1, 256, 0, stream>>>(bn_acc);
  k_bn_stats<<<256, 256, 0, stream>>>(hB, NN, bn_acc);
  k_bn_finalize<<<1, 128, 0, stream>>>(bn_acc, bn_st, NN);
  k_bn_apply_relu<<<ELEMG, 256, 0, stream>>>(hB, bn_st, g2, be2, NN);

  // ---- attention layers ----
  const float* cur = hB;
  for (int l = 0; l < 2; ++l) {
    k_gemm<128, false><<<GEMMG, 256, 0, stream>>>(cur, wt_q[l], bq + l * 128, Qb, NN);
    k_gemm<128, false><<<GEMMG, 256, 0, stream>>>(cur, wt_k[l], bk + l * 128, Kb, NN);
    k_gemm<128, false><<<GEMMG, 256, 0, stream>>>(cur, wt_v[l], bv + l * 128, Vb, NN);

    k_seg_init<<<SEGG, 256, 0, stream>>>(m_enc, ssum, asum);
    k_scores<<<EDGEW, 256, 0, stream>>>(src, dst, Qb, Kb, scores, m_enc);
    k_exp_sum<<<EDGEG4, 256, 0, stream>>>(dst, scores, m_enc, ssum);
    k_norm_acc<<<EDGEG4, 256, 0, stream>>>(dst, scores, ssum, asum);
    k_scale_v<<<ELEMG, 256, 0, stream>>>(Vb, asum, Qb);

    float* outp = (l == 1) ? (float*)d_out : hA;
    k_gemm<128, true><<<GEMMG, 256, 0, stream>>>(Qb, wt_o[l], bo + l * 128, outp, NN);
    cur = outp;
  }
}
